// FrameConsistencyLoss_72550587564373
// MI455X (gfx1250) — matrix operand order repaired and checked by reading, not compiled, not run on hardware
//
#include <hip/hip_runtime.h>
#include <hip/hip_bf16.h>
#include <stdint.h>

// ---------------------------------------------------------------------------
// FrameConsistencyLoss on MI455X (gfx1250, wave32, WMMA)
//
// diff = x_a @ W[id_a] + b[id_a] - (x_b @ W[id_b] + b[id_b]) ; loss = mean(diff^2)
//
// One GEMM per 16-row tile via an augmented K=128 axis:
//   k in [id_a*14, id_a*14+14)  -> +x_a      (expert dispatch a)
//   k == 56 + id_a              -> +1         (bias one-hot a)
//   k in [64+id_b*14, ...+14)   -> -x_b      (expert dispatch b)
//   k == 64 + 56 + id_b         -> -1         (bias one-hot b)
// B (64x64) = [W[0..3] stacked; b[0..3]; 0pad], reused for both K halves.
// f32 precision recovered by hi/lo f16 splitting: ah*bh + ah*bl + al*bh.
// Separate hi/lo f16 LDS planes -> fragment loads are 2x ds_load_b128 each.
// ---------------------------------------------------------------------------

typedef __attribute__((ext_vector_type(16))) _Float16 v16h;
typedef __attribute__((ext_vector_type(8)))  _Float16 v8h;
typedef __attribute__((ext_vector_type(8)))  float    v8f;

#define REL_DIM 14
#define CAN_DIM 64
#define N_TYPES 4
#define KAUG    64                 // per-side augmented K (56 W + 4 one-hot + 4 pad)
#define WAVES_PER_BLOCK 4
#define THREADS (WAVES_PER_BLOCK * 32)
#define MAX_BLOCKS 2048

static __device__ inline v16h cat16(v8h a, v8h b) {
  return __builtin_shufflevector(a, b, 0,1,2,3,4,5,6,7,8,9,10,11,12,13,14,15);
}

__global__ __launch_bounds__(THREADS)
void moe_diff_mse_kernel(const float* __restrict__ xa, const float* __restrict__ xb,
                         const int*  __restrict__ ida, const int*  __restrict__ idb,
                         const float* __restrict__ W,  const float* __restrict__ bias,
                         float* __restrict__ blockPartials, int nRows)
{
  // B planes stored TRANSPOSED [n][k] so a B fragment is 16 contiguous halves.
  __shared__ _Float16 Bthi[CAN_DIM][KAUG];                     // 8 KB
  __shared__ _Float16 Btlo[CAN_DIM][KAUG];                     // 8 KB
  // Per-wave augmented A tile, separate hi/lo planes, [m][k] contiguous in k.
  __shared__ _Float16 Ahi[WAVES_PER_BLOCK][16][2 * KAUG];      // 16 KB
  __shared__ _Float16 Alo[WAVES_PER_BLOCK][16][2 * KAUG];      // 16 KB
  __shared__ float    red[WAVES_PER_BLOCK];

  const int tid  = threadIdx.x;
  const int lane = tid & 31;
  const int wave = tid >> 5;

  // ---- build augmented B (stacked W, bias rows, zero pad), split hi/lo ----
  for (int idx = tid; idx < KAUG * CAN_DIM; idx += THREADS) {
    int n = idx >> 6, k = idx & 63;
    float v = 0.f;
    if (k < N_TYPES * REL_DIM)                v = W[k * CAN_DIM + n];   // k == e*14+r
    else if (k < N_TYPES * REL_DIM + N_TYPES) v = bias[(k - N_TYPES * REL_DIM) * CAN_DIM + n];
    _Float16 h = (_Float16)v;
    Bthi[n][k] = h;
    Btlo[n][k] = (_Float16)(v - (float)h);
  }
  __syncthreads();

  // ---- resident B fragments: 2 k-steps x 4 n-tiles, hi & lo --------------
  // B frag (32x16 f16, KxN): lane L -> n = L&15 ; element i -> k = (L>>4)*16 + i
  v16h bhi[2][4], blo[2][4];
  {
    const int n_lane = lane & 15;
    const int kb     = (lane >> 4) * 16;
#pragma unroll
    for (int ks = 0; ks < 2; ++ks)
#pragma unroll
      for (int nt = 0; nt < 4; ++nt) {
        const v8h* ph = (const v8h*)&Bthi[nt * 16 + n_lane][ks * 32 + kb];
        const v8h* pl = (const v8h*)&Btlo[nt * 16 + n_lane][ks * 32 + kb];
        bhi[ks][nt] = cat16(ph[0], ph[1]);
        blo[ks][nt] = cat16(pl[0], pl[1]);
      }
  }

  // ---- per-wave tile loop --------------------------------------------------
  const int m    = lane & 15;            // row within tile (A frag: M = lane&15)
  const int side = lane >> 4;            // scatter role: 0 = +a rows, 1 = -b rows
  const int kloc = (lane >> 4) * 16;     // A frag K base within 32 (contiguous, matches B)

  const int nTiles = (nRows + 15) >> 4;
  const int gwave  = blockIdx.x * WAVES_PER_BLOCK + wave;
  const int nWaves = gridDim.x * WAVES_PER_BLOCK;

  float lsum = 0.f;
  _Float16 (*Ahi_w)[2 * KAUG] = Ahi[wave];
  _Float16 (*Alo_w)[2 * KAUG] = Alo[wave];

  for (int t = gwave; t < nTiles; t += nWaves) {
    asm volatile("" ::: "memory");
    // zero both planes: each 16*128 halves = 4 KB = 256 uint4 -> 8 per lane
    uint4* zh = (uint4*)&Ahi_w[0][0];
    uint4* zl = (uint4*)&Alo_w[0][0];
#pragma unroll
    for (int j = 0; j < 8; ++j) {
      zh[lane + 32 * j] = make_uint4(0u, 0u, 0u, 0u);
      zl[lane + 32 * j] = make_uint4(0u, 0u, 0u, 0u);
    }

    // scatter this lane's row (sign folds the a-b subtraction into the GEMM)
    int r = t * 16 + m;
    if (r < nRows) {
      const float* rowp = (side ? xb : xa) + (size_t)r * REL_DIM;
      int   e    = (side ? idb : ida)[r];
      float sgn  = side ? -1.f : 1.f;
      int   col0 = side * KAUG + e * REL_DIM;
#pragma unroll
      for (int j = 0; j < REL_DIM; ++j) {
        float v = sgn * rowp[j];
        _Float16 h = (_Float16)v;
        Ahi_w[m][col0 + j] = h;
        Alo_w[m][col0 + j] = (_Float16)(v - (float)h);
      }
      Ahi_w[m][side * KAUG + N_TYPES * REL_DIM + e] = (_Float16)sgn;  // bias one-hot (exact)
    }
    asm volatile("" ::: "memory");   // same-wave DS is HW in-order; block IR reordering

    v8f acc[4];
#pragma unroll
    for (int nt = 0; nt < 4; ++nt) acc[nt] = v8f{0.f,0.f,0.f,0.f,0.f,0.f,0.f,0.f};

#pragma unroll
    for (int ks = 0; ks < 4; ++ks) {
      // A fragment (16x32 f16, contiguous k = 16h + i): elements 0-15 are
      // 16 contiguous halves -> two ds_load_b128 per plane, no half-splicing.
      const int base0 = ks * 32 + kloc;
      const v8h* ph = (const v8h*)&Ahi_w[m][base0];
      const v8h* pl = (const v8h*)&Alo_w[m][base0];
      v16h ahi = cat16(ph[0], ph[1]);     // halves [base0..+15]
      v16h alo = cat16(pl[0], pl[1]);

      const int bks = ks & 1;   // B rows repeat for the (negated) b-side K half
#pragma unroll
      for (int nt = 0; nt < 4; ++nt) {
        acc[nt] = __builtin_amdgcn_wmma_f32_16x16x32_f16(false, ahi, false, bhi[bks][nt],
                                                         (short)0, acc[nt], false, false);
        acc[nt] = __builtin_amdgcn_wmma_f32_16x16x32_f16(false, ahi, false, blo[bks][nt],
                                                         (short)0, acc[nt], false, false);
        acc[nt] = __builtin_amdgcn_wmma_f32_16x16x32_f16(false, alo, false, bhi[bks][nt],
                                                         (short)0, acc[nt], false, false);
      }
    }

    // acc == proj_a - proj_b for this 16x64 tile; square-accumulate
#pragma unroll
    for (int nt = 0; nt < 4; ++nt)
#pragma unroll
      for (int j = 0; j < 8; ++j)
        lsum += acc[nt][j] * acc[nt][j];
  }

  // wave32 butterfly reduce, then block reduce (deterministic, no fp atomics)
#pragma unroll
  for (int off = 16; off > 0; off >>= 1)
    lsum += __shfl_xor(lsum, off, 32);
  if (lane == 0) red[wave] = lsum;
  __syncthreads();
  if (tid == 0) {
    float s = 0.f;
    for (int wv = 0; wv < WAVES_PER_BLOCK; ++wv) s += red[wv];
    blockPartials[blockIdx.x] = s;
  }
}

__global__ void reduce_kernel(const float* __restrict__ partials, int n,
                              float* __restrict__ out, float invCount)
{
  __shared__ float red[256];
  float s = 0.f;
  for (int i = threadIdx.x; i < n; i += 256) s += partials[i];
  red[threadIdx.x] = s;
  __syncthreads();
  for (int st = 128; st > 0; st >>= 1) {
    if (threadIdx.x < st) red[threadIdx.x] += red[threadIdx.x + st];
    __syncthreads();
  }
  if (threadIdx.x == 0) out[0] = red[0] * invCount;
}

extern "C" void kernel_launch(void* const* d_in, const int* in_sizes, int n_in,
                              void* d_out, int out_size, void* d_ws, size_t ws_size,
                              hipStream_t stream) {
  const float* xa  = (const float*)d_in[0];
  const float* xb  = (const float*)d_in[1];
  const int*   ia  = (const int*)d_in[2];
  const int*   ib  = (const int*)d_in[3];
  const float* W   = (const float*)d_in[4];
  const float* b   = (const float*)d_in[5];
  const int nRows  = in_sizes[2];                    // N
  float* partials  = (float*)d_ws;

  int nTiles = (nRows + 15) / 16;
  int blocks = MAX_BLOCKS;
  int maxB   = (nTiles + WAVES_PER_BLOCK - 1) / WAVES_PER_BLOCK;
  if (blocks > maxB) blocks = maxB;
  if (blocks < 1)    blocks = 1;

  moe_diff_mse_kernel<<<blocks, THREADS, 0, stream>>>(xa, xb, ia, ib, W, b, partials, nRows);

  double cnt = (double)nRows * (double)CAN_DIM;
  reduce_kernel<<<1, 256, 0, stream>>>(partials, blocks, (float*)d_out, (float)(1.0 / cnt));
}
